// network_23038204576102
// MI455X (gfx1250) — matrix operand order repaired and checked by reading, not compiled, not run on hardware
//
#include <hip/hip_runtime.h>

typedef __attribute__((ext_vector_type(16))) _Float16 v16h;
typedef __attribute__((ext_vector_type(8)))  float    v8f;
typedef __attribute__((ext_vector_type(4)))  float    v4f;

#define N_NODES   100000
#define N_EDGES   200000
#define N_GRAPHS  4096
#define NODE_DIM  64
#define HID       256
#define N_GAT     4
#define NEG_SLOPE 0.2f
#define EPS_F     1e-9f

// ---------------------------------------------------------------------------
// Weight prep: Wt[n*K + k] = (f16) W[k*N + n]   (transpose + f32->f16 convert)
// ---------------------------------------------------------------------------
__global__ void k_transpose_f32_to_f16(const float* __restrict__ W,
                                       _Float16* __restrict__ Wt, int K, int N) {
  int total = K * N;
  for (int i = blockIdx.x * blockDim.x + threadIdx.x; i < total;
       i += gridDim.x * blockDim.x) {
    int n = i % N, k = i / N;
    Wt[n * K + k] = (_Float16)W[k * N + n];
  }
}

// ---------------------------------------------------------------------------
// WMMA GEMM: C[M x (NTILES*16)] = A[M x K](f32) * B[K x (NTILES*16)]
// Bt is pre-transposed f16: Bt[n*K + k] (K contiguous -> 32B/lane loads).
//
// Register strategy: one wave owns a 16-row strip across all output columns.
// ALL K fragments of A are preloaded (KFRAGS * 8 VGPRs) so A is read from
// HBM exactly once; output tiles are produced two at a time so only 2
// accumulators + 2 B fragments are ever live (no spills, two independent
// WMMA chains to hide latency behind L2-hot B loads). HAS_BIAS/RELU are
// compile-time so the hot (no-bias) instantiation has a branch-free loop.
// ---------------------------------------------------------------------------
template <int KFRAGS, int NTILES, bool HAS_BIAS, bool RELU>
__global__ void __launch_bounds__(256)
k_wmma_gemm_rowstrip(const float* __restrict__ A,
                     const _Float16* __restrict__ Bt,
                     const float* __restrict__ bias,
                     float* __restrict__ C, int M) {
  constexpr int K = KFRAGS * 32;
  constexpr int N = NTILES * 16;

  const int lane   = threadIdx.x & 31;
  const int wave   = threadIdx.x >> 5;
  const int tileM  = blockIdx.x * (blockDim.x >> 5) + wave;
  const int mtiles = M >> 4;
  if (tileM >= mtiles) return;               // uniform per wave (EXEC all-1s)

  const int hsel = lane >> 4;                // 0: lanes 0-15, 1: lanes 16-31
  const int l16  = lane & 15;

  // ---- preload + convert all A fragments (contiguous k order used here, not the ISA table order:
  //      lane = M, lanes<16 hold K 0..15, lanes>=16 hold K 16..31)
  const int    row  = tileM * 16 + l16;
  const float* arow = A + (long)row * K + hsel * 16;
  v16h a[KFRAGS];
#pragma unroll
  for (int f = 0; f < KFRAGS; ++f) {
    const float* ap = arow + f * 32;
    v8f lo = *(const v8f*)(ap);              // K: base .. base+7
    v8f hi = *(const v8f*)(ap + 8);          // K: base+8 .. base+15
#pragma unroll
    for (int i = 0; i < 8; ++i) {
      a[f][i]     = (_Float16)lo[i];
      a[f][8 + i] = (_Float16)hi[i];
    }
  }

  // ---- B fragment base (B uses the same contiguous k order as A: lane = N,
  //      lanes<16 hold K 0..15, lanes>=16 hold K 16..31)
  const _Float16* bbase = Bt + (long)l16 * K + hsel * 16;
  const long      bcolstride = (long)16 * K; // halfs between adjacent N tiles

#pragma unroll 1
  for (int n = 0; n < NTILES; n += 2) {
    v8f acc0, acc1;
#pragma unroll
    for (int i = 0; i < 8; ++i) { acc0[i] = 0.0f; acc1[i] = 0.0f; }

    const _Float16* bp0 = bbase + (long)n * bcolstride;
    const _Float16* bp1 = bp0 + bcolstride;
#pragma unroll
    for (int f = 0; f < KFRAGS; ++f) {
      v16h b0 = *(const v16h*)(bp0 + f * 32);
      v16h b1 = *(const v16h*)(bp1 + f * 32);
      acc0 = __builtin_amdgcn_wmma_f32_16x16x32_f16(
          false, a[f], false, b0, (short)0, acc0, false, false);
      acc1 = __builtin_amdgcn_wmma_f32_16x16x32_f16(
          false, a[f], false, b1, (short)0, acc1, false, false);
    }

    // store: D layout -> VGPR v holds M = v + 8*hsel, lane%16 = N
    const int col0 = n * 16 + l16;
    float bv0 = 0.0f, bv1 = 0.0f;
    if (HAS_BIAS) { bv0 = bias[col0]; bv1 = bias[col0 + 16]; }
    float* crow = C + (long)(tileM * 16 + hsel * 8) * N + col0;
#pragma unroll
    for (int v = 0; v < 8; ++v) {
      float v0 = acc0[v] + bv0;
      float v1 = acc1[v] + bv1;
      if (RELU) { v0 = fmaxf(v0, 0.0f); v1 = fmaxf(v1, 0.0f); }
      crow[(long)v * N]      = v0;
      crow[(long)v * N + 16] = v1;
    }
  }
}

// ---------------------------------------------------------------------------
// el[i] = z[i,:] . al ; er[i] = z[i,:] . ar   (one wave per node)
// ---------------------------------------------------------------------------
__global__ void k_attn_scores(const float* __restrict__ z,
                              const float* __restrict__ al,
                              const float* __restrict__ ar,
                              float* __restrict__ el, float* __restrict__ er,
                              int N) {
  int lane = threadIdx.x & 31;
  int node = blockIdx.x * (blockDim.x >> 5) + (threadIdx.x >> 5);
  if (node >= N) return;
  const float* zr = z + (long)node * HID;
  int c = lane * 8;
  v4f z0 = *(const v4f*)(zr + c), z1 = *(const v4f*)(zr + c + 4);
  v4f a0 = *(const v4f*)(al + c), a1 = *(const v4f*)(al + c + 4);
  v4f r0 = *(const v4f*)(ar + c), r1 = *(const v4f*)(ar + c + 4);
  float sl = 0.f, sr = 0.f;
#pragma unroll
  for (int i = 0; i < 4; ++i) {
    sl += z0[i] * a0[i] + z1[i] * a1[i];
    sr += z0[i] * r0[i] + z1[i] * r1[i];
  }
  for (int off = 16; off > 0; off >>= 1) {
    sl += __shfl_xor(sl, off, 32);
    sr += __shfl_xor(sr, off, 32);
  }
  if (lane == 0) { el[node] = sl; er[node] = sr; }
}

__global__ void k_init_attn(float* __restrict__ emax, float* __restrict__ esum,
                            int N) {
  int i = blockIdx.x * blockDim.x + threadIdx.x;
  if (i < N) { emax[i] = -__builtin_inff(); esum[i] = 0.0f; }
}

__device__ __forceinline__ void atomicMaxF32(float* addr, float val) {
  unsigned int* ua  = (unsigned int*)addr;
  unsigned int  old = *ua;
  while (__uint_as_float(old) < val) {
    unsigned int assumed = old;
    old = atomicCAS(ua, assumed, __float_as_uint(val));
    if (old == assumed) break;
  }
}

__global__ void k_edge_max(const float* __restrict__ el,
                           const float* __restrict__ er,
                           const int* __restrict__ src,
                           const int* __restrict__ dst,
                           float* __restrict__ e, float* __restrict__ emax,
                           int E) {
  int j = blockIdx.x * blockDim.x + threadIdx.x;
  if (j >= E) return;
  float v = el[src[j]] + er[dst[j]];
  v = (v > 0.f) ? v : NEG_SLOPE * v;         // leaky relu
  e[j] = v;
  atomicMaxF32(&emax[dst[j]], v);
}

__global__ void k_edge_exp(const float* __restrict__ e,
                           const int* __restrict__ dst,
                           const float* __restrict__ emax,
                           float* __restrict__ ex, float* __restrict__ esum,
                           int E) {
  int j = blockIdx.x * blockDim.x + threadIdx.x;
  if (j >= E) return;
  float mx = emax[dst[j]];
  if (!__builtin_isfinite(mx)) mx = 0.0f;
  float v = expf(e[j] - mx);
  ex[j] = v;
  atomicAdd(&esum[dst[j]], v);
}

// m[dst,:] += (ex/(esum[dst]+eps)) * z[src,:]  -- 64 threads per edge, f32x4
__global__ void k_scatter_msg(const float* __restrict__ z,
                              const int* __restrict__ src,
                              const int* __restrict__ dst,
                              const float* __restrict__ ex,
                              const float* __restrict__ esum,
                              float* __restrict__ m, int E) {
  int epb = blockDim.x >> 6;
  int j   = blockIdx.x * epb + (threadIdx.x >> 6);
  if (j >= E) return;
  int   t     = threadIdx.x & 63;
  int   s     = src[j], d = dst[j];
  float alpha = ex[j] / (esum[d] + EPS_F);
  v4f zv = *(const v4f*)(z + (long)s * HID + t * 4);
  float* mp = m + (long)d * HID + t * 4;
#pragma unroll
  for (int i = 0; i < 4; ++i) atomicAdd(mp + i, alpha * zv[i]);
}

__global__ void k_resid_relu(float* __restrict__ h, const float* __restrict__ m,
                             long total4) {
  v4f*       h4 = (v4f*)h;
  const v4f* m4 = (const v4f*)m;
  for (long i = blockIdx.x * (long)blockDim.x + threadIdx.x; i < total4;
       i += (long)gridDim.x * blockDim.x) {
    v4f a = h4[i], b = m4[i];
#pragma unroll
    for (int k = 0; k < 4; ++k) {
      float v = a[k] + b[k];
      a[k] = v > 0.f ? v : 0.f;
    }
    h4[i] = a;
  }
}

__global__ void k_pool(const float* __restrict__ h, const int* __restrict__ gid,
                       float* __restrict__ g, int N) {
  int npb  = blockDim.x >> 6;
  int node = blockIdx.x * npb + (threadIdx.x >> 6);
  if (node >= N) return;
  int t  = threadIdx.x & 63;
  int gg = gid[node];
  v4f hv = *(const v4f*)(h + (long)node * HID + t * 4);
  float* gp = g + (long)gg * HID + t * 4;
#pragma unroll
  for (int i = 0; i < 4; ++i) atomicAdd(gp + i, hv[i]);
}

// out[row] = g2[row,:] . w3 + b3   (one wave per graph, 64-dim dot)
__global__ void k_final(const float* __restrict__ g2,
                        const float* __restrict__ w3,
                        const float* __restrict__ b3, float* __restrict__ out,
                        int G) {
  int lane = threadIdx.x & 31;
  int row  = blockIdx.x * (blockDim.x >> 5) + (threadIdx.x >> 5);
  if (row >= G) return;
  const float* gr = g2 + (long)row * 64;
  float s = gr[lane * 2] * w3[lane * 2] + gr[lane * 2 + 1] * w3[lane * 2 + 1];
  for (int off = 16; off > 0; off >>= 1) s += __shfl_xor(s, off, 32);
  if (lane == 0) out[row] = s + b3[0];
}

// ---------------------------------------------------------------------------
extern "C" void kernel_launch(void* const* d_in, const int* in_sizes, int n_in,
                              void* d_out, int out_size, void* d_ws,
                              size_t ws_size, hipStream_t stream) {
  const float* node_feats = (const float*)d_in[0];
  const int*   src        = (const int*)d_in[1];
  const int*   dst        = (const int*)d_in[2];
  const int*   graph_ids  = (const int*)d_in[3];
  const float* w_embed    = (const float*)d_in[4];
  const float* b_embed    = (const float*)d_in[5];
  const float* gat_W      = (const float*)d_in[6];
  const float* gat_al     = (const float*)d_in[7];
  const float* gat_ar     = (const float*)d_in[8];
  const float* w1         = (const float*)d_in[9];
  const float* b1         = (const float*)d_in[10];
  const float* w2         = (const float*)d_in[11];
  const float* b2         = (const float*)d_in[12];
  const float* w3         = (const float*)d_in[13];
  const float* b3         = (const float*)d_in[14];
  float*       out        = (float*)d_out;

  // workspace carve-out
  char* p = (char*)d_ws;
  auto  carve = [&](size_t bytes) {
    char* q = p;
    p += (bytes + 255) & ~(size_t)255;
    return q;
  };
  float* h    = (float*)carve((size_t)N_NODES * HID * 4);
  float* z    = (float*)carve((size_t)N_NODES * HID * 4);
  float* m    = (float*)carve((size_t)N_NODES * HID * 4);
  float* el   = (float*)carve((size_t)N_NODES * 4);
  float* er   = (float*)carve((size_t)N_NODES * 4);
  float* emax = (float*)carve((size_t)N_NODES * 4);
  float* esum = (float*)carve((size_t)N_NODES * 4);
  float* ebuf = (float*)carve((size_t)N_EDGES * 4);
  float* exb  = (float*)carve((size_t)N_EDGES * 4);
  float* g    = (float*)carve((size_t)N_GRAPHS * HID * 4);
  float* g1   = (float*)carve((size_t)N_GRAPHS * 128 * 4);
  float* g2   = (float*)carve((size_t)N_GRAPHS * 64 * 4);
  _Float16* wt_embed = (_Float16*)carve((size_t)HID * NODE_DIM * 2);
  _Float16* wtg      = (_Float16*)carve((size_t)N_GAT * HID * HID * 2);
  _Float16* w1t      = (_Float16*)carve((size_t)128 * HID * 2);
  _Float16* w2t      = (_Float16*)carve((size_t)64 * 128 * 2);

  // weight transpose + f16 convert (tiny, L2-resident afterwards)
  k_transpose_f32_to_f16<<<64, 256, 0, stream>>>(w_embed, wt_embed, NODE_DIM, HID);
  for (int l = 0; l < N_GAT; ++l)
    k_transpose_f32_to_f16<<<256, 256, 0, stream>>>(gat_W + (size_t)l * HID * HID,
                                                    wtg + (size_t)l * HID * HID,
                                                    HID, HID);
  k_transpose_f32_to_f16<<<128, 256, 0, stream>>>(w1, w1t, HID, 128);
  k_transpose_f32_to_f16<<<32, 256, 0, stream>>>(w2, w2t, 128, 64);

  const int WPB        = 8;                      // waves (16-row strips) / block
  const int mt_nodes   = N_NODES / 16;           // 6250
  const int gemmBlocks = (mt_nodes + WPB - 1) / WPB;

  // embed: h = node_feats @ w_embed + b_embed   (K=64 -> 2 K-frags)
  k_wmma_gemm_rowstrip<2, 16, true, false><<<gemmBlocks, 256, 0, stream>>>(
      node_feats, wt_embed, b_embed, h, N_NODES);

  for (int l = 0; l < N_GAT; ++l) {
    k_wmma_gemm_rowstrip<8, 16, false, false><<<gemmBlocks, 256, 0, stream>>>(
        h, wtg + (size_t)l * HID * HID, nullptr, z, N_NODES);
    k_attn_scores<<<(N_NODES + 7) / 8, 256, 0, stream>>>(
        z, gat_al + l * HID, gat_ar + l * HID, el, er, N_NODES);
    k_init_attn<<<(N_NODES + 255) / 256, 256, 0, stream>>>(emax, esum, N_NODES);
    hipMemsetAsync(m, 0, (size_t)N_NODES * HID * 4, stream);
    k_edge_max<<<(N_EDGES + 255) / 256, 256, 0, stream>>>(el, er, src, dst,
                                                          ebuf, emax, N_EDGES);
    k_edge_exp<<<(N_EDGES + 255) / 256, 256, 0, stream>>>(ebuf, dst, emax, exb,
                                                          esum, N_EDGES);
    k_scatter_msg<<<(N_EDGES + 3) / 4, 256, 0, stream>>>(z, src, dst, exb,
                                                         esum, m, N_EDGES);
    k_resid_relu<<<4096, 256, 0, stream>>>(h, m, (long)N_NODES * HID / 4);
  }

  // sum pooling per graph
  hipMemsetAsync(g, 0, (size_t)N_GRAPHS * HID * 4, stream);
  k_pool<<<(N_NODES + 3) / 4, 256, 0, stream>>>(h, graph_ids, g, N_NODES);

  // MLP head
  const int mt_g = N_GRAPHS / 16;  // 256
  k_wmma_gemm_rowstrip<8, 8, true, true><<<(mt_g + WPB - 1) / WPB, 256, 0, stream>>>(
      g, w1t, b1, g1, N_GRAPHS);
  k_wmma_gemm_rowstrip<4, 4, true, true><<<(mt_g + WPB - 1) / WPB, 256, 0, stream>>>(
      g1, w2t, b2, g2, N_GRAPHS);
  k_final<<<(N_GRAPHS + 7) / 8, 256, 0, stream>>>(g2, w3, b3, out, N_GRAPHS);
}
